// EdgeTamMemoryAttentionLayer_12936441496039
// MI455X (gfx1250) — matrix operand order repaired and checked by reading, not compiled, not run on hardware
//
#include <hip/hip_runtime.h>
#include <math.h>

typedef __attribute__((ext_vector_type(16))) _Float16 v16h;
typedef __attribute__((ext_vector_type(8)))  _Float16 v8h;
typedef __attribute__((ext_vector_type(8)))  float    v8f;

#define WMMA_F16(A,B,C) __builtin_amdgcn_wmma_f32_16x16x32_f16(false,(A),false,(B),(short)0,(C),false,false)

#define SQ      4096
#define DM      256
#define SKV     28700
#define SKVP    28704
#define KVD     64
#define FFD     2048
#define PERFRM  4100   // 4096 rope rows + 4 no-rope rows per frame
#define LN_EPS  1e-5f
#define SPLITS  8      // flash-decoding KV splits for attention occupancy

// ---------------------------------------------------------------------------
// Fragment loaders (wave32 WMMA f16; A and B use the same contiguous k order of this file's choosing)
// A 16x32 f16: lane l -> row m=l&15; halves = K {0-15} (l<16) / {16-31}
// B 32x16 f16: lane l -> col n=l&15; halves = K {0-15} (l<16) / {16-31}
// C 16x16 f32: lane l -> col n=l&15; vgpr j -> row m = j + (l<16?0:8)
// ---------------------------------------------------------------------------
static __device__ inline v16h load_a_frag(const _Float16* __restrict__ base,
                                          int stride, int m, int kk, int lane) {
  int kofs = (lane < 16) ? 0 : 16;
  const _Float16* p = base + (size_t)m * stride + kk + kofs;
  v8h lo = *(const v8h*)p;
  v8h hi = *(const v8h*)(p + 8);
  v16h r;
#pragma unroll
  for (int i = 0; i < 8; ++i) { r[i] = lo[i]; r[i + 8] = hi[i]; }
  return r;
}

static __device__ inline v16h load_b_frag(const _Float16* __restrict__ base,
                                          int stride, int n, int kk, int lane) {
  int kofs = (lane < 16) ? 0 : 16;
  return *(const v16h*)(base + (size_t)n * stride + kk + kofs);
}

// ---------------------------------------------------------------------------
// Small utility kernels
// ---------------------------------------------------------------------------
__global__ void k_copy_f32(const float* __restrict__ src, float* __restrict__ dst, int n) {
  int i = blockIdx.x * blockDim.x + threadIdx.x;
  if (i < n) dst[i] = src[i];
}

// W[K x N] f32 -> Wt[N x K] f16
__global__ void k_wt(const float* __restrict__ W, _Float16* __restrict__ Wt, int K, int N) {
  int i = blockIdx.x * blockDim.x + threadIdx.x;
  if (i >= K * N) return;
  int k = i / N, n = i % N;
  Wt[(size_t)n * K + k] = (_Float16)W[i];
}

// kin16 = f16(keys + kpe), keys16 = f16(keys); zero the 4 padding rows
__global__ void k_addcvt(const float* __restrict__ keys, const float* __restrict__ kpe,
                         _Float16* __restrict__ kin16, _Float16* __restrict__ keys16) {
  int i = blockIdx.x * blockDim.x + threadIdx.x;
  if (i >= SKVP * KVD) return;
  int row = i >> 6;
  if (row < SKV) {
    kin16[i]  = (_Float16)(keys[i] + kpe[i]);
    keys16[i] = (_Float16)keys[i];
  } else {
    kin16[i]  = (_Float16)0.0f;
    keys16[i] = (_Float16)0.0f;
  }
}

// LayerNorm over 256 cols, f32 in -> f16 out. One row per block of 256.
__global__ __launch_bounds__(256) void k_ln(const float* __restrict__ x,
                                            const float* __restrict__ w,
                                            const float* __restrict__ b,
                                            _Float16* __restrict__ out) {
  __shared__ float red[256];
  int row = blockIdx.x, t = threadIdx.x;
  float v = x[(size_t)row * DM + t];
  red[t] = v; __syncthreads();
#pragma unroll
  for (int s = 128; s > 0; s >>= 1) { if (t < s) red[t] += red[t + s]; __syncthreads(); }
  float mu = red[0] * (1.0f / DM); __syncthreads();
  float d = v - mu;
  red[t] = d * d; __syncthreads();
#pragma unroll
  for (int s = 128; s > 0; s >>= 1) { if (t < s) red[t] += red[t + s]; __syncthreads(); }
  float var = red[0] * (1.0f / DM);
  out[(size_t)row * DM + t] = (_Float16)(d * rsqrtf(var + LN_EPS) * w[t] + b[t]);
}

// In-place 2D RoPE on f16 [rows x 256]; cos/sin are [4096 x 128] f32.
// mode 0: table row = row.  mode 1: 7 frames of 4100 rows, first 4 rows no-rope.
__global__ void k_rope(_Float16* __restrict__ x, const float* __restrict__ cosT,
                       const float* __restrict__ sinT, int mode) {
  int row = blockIdx.x, t = threadIdx.x;   // t in [0,128)
  int tr;
  if (mode == 1) {
    if (row >= SKV) return;
    int rif = row % PERFRM;
    if (rif < 4) return;                   // no-rope rows pass through
    tr = rif - 4;
  } else {
    tr = row;
  }
  size_t base = (size_t)row * DM;
  float xa = (float)x[base + t];
  float xb = (float)x[base + t + 128];
  int ci = tr * 128 + (t >> 1);
  float ca = cosT[ci],      sa = sinT[ci];
  float cb = cosT[ci + 64], sb = sinT[ci + 64];
  x[base + t]       = (_Float16)(xa * ca - xb * sa);
  x[base + t + 128] = (_Float16)(xb * cb + xa * sb);
}

// ---------------------------------------------------------------------------
// WMMA GEMM, 32x32 tile per wave (2x2 fragments, 4 wmma per K-step):
//   C[M x N] = A[M x K](f16) @ Bt^T  (Bt is [N x K] f16) + bias
// OUT_MODE 0: f16 row-major out.  1: f16 transposed out (out[n*tstride+m], for V^T).
//          2: accumulate f32 into resid[m*N+n].
// ---------------------------------------------------------------------------
template <int OUT_MODE, bool RELU>
__global__ __launch_bounds__(128) void
k_gemm(const _Float16* __restrict__ A, const _Float16* __restrict__ Bt,
       const float* __restrict__ bias, _Float16* __restrict__ out16,
       float* __restrict__ resid, int M, int N, int K, int tstride) {
  int lane = threadIdx.x & 31;
  int wid  = (blockIdx.x * blockDim.x + threadIdx.x) >> 5;
  int ntn = N >> 5, ntm = M >> 5;
  if (wid >= ntm * ntn) return;
  int mt = wid / ntn, nt = wid % ntn;
  int mbase = mt << 5, nbase = nt << 5;

  v8f acc[2][2] = {};
  for (int kk = 0; kk < K; kk += 32) {
    v16h a0 = load_a_frag(A, K, mbase + (lane & 15), kk, lane);
    v16h a1 = load_a_frag(A, K, mbase + 16 + (lane & 15), kk, lane);
    v16h b0 = load_b_frag(Bt, K, nbase + (lane & 15), kk, lane);
    v16h b1 = load_b_frag(Bt, K, nbase + 16 + (lane & 15), kk, lane);
    acc[0][0] = WMMA_F16(a0, b0, acc[0][0]);
    acc[0][1] = WMMA_F16(a0, b1, acc[0][1]);
    acc[1][0] = WMMA_F16(a1, b0, acc[1][0]);
    acc[1][1] = WMMA_F16(a1, b1, acc[1][1]);
  }

  int rofs = (lane < 16) ? 0 : 8;
#pragma unroll
  for (int mi = 0; mi < 2; ++mi) {
#pragma unroll
    for (int ni = 0; ni < 2; ++ni) {
      int n = nbase + ni * 16 + (lane & 15);
      float bv = bias[n];
      if (OUT_MODE == 1) {
        v8h st;
#pragma unroll
        for (int j = 0; j < 8; ++j) {
          float v = acc[mi][ni][j] + bv;
          if (RELU) v = fmaxf(v, 0.0f);
          st[j] = (_Float16)v;
        }
        *(v8h*)(out16 + (size_t)n * tstride + mbase + mi * 16 + rofs) = st;
      } else {
#pragma unroll
        for (int j = 0; j < 8; ++j) {
          int m = mbase + mi * 16 + rofs + j;
          float v = acc[mi][ni][j] + bv;
          if (RELU) v = fmaxf(v, 0.0f);
          if (OUT_MODE == 0) out16[(size_t)m * N + n] = (_Float16)v;
          if (OUT_MODE == 2) resid[(size_t)m * N + n] += v;
        }
      }
    }
  }
}

// ---------------------------------------------------------------------------
// Flash attention with KV-splitting (flash-decoding). One wave per
// (16-query tile, KV split). blockIdx.y = split, each split covers `cpw`
// 32-key chunks. Emits locally-normalized partial O plus (m, l) per query.
//   S^T = K_tile(16x256) @ Q^T       -> lane owns one query column
//   O^T += V^T(16x32) @ P^T(32x16)   -> P^T B-frag built with shfl_xor(16)
// q:[4096x256] k:[nchunks*32 x 256] vt:[256 x vts], all f16.
// opart:[SPLITS x 4096 x 256] f16, mbuf/lbuf:[SPLITS x 4096] f32.
// ---------------------------------------------------------------------------
__global__ __launch_bounds__(128) void
k_attn(const _Float16* __restrict__ q, const _Float16* __restrict__ k,
       const _Float16* __restrict__ vt, _Float16* __restrict__ opart,
       float* __restrict__ mbuf, float* __restrict__ lbuf,
       int skv, int vts, int nchunks, int cpw) {
  int lane = threadIdx.x & 31;
  int wave = (blockIdx.x * blockDim.x + threadIdx.x) >> 5;
  if (wave >= (SQ >> 4)) return;
  int split = blockIdx.y;
  int c0 = split * cpw;
  int c1 = c0 + cpw; if (c1 > nchunks) c1 = nchunks;
  int qbase = wave << 4;
  int qcol = qbase + (lane & 15);
  int krow = (lane < 16) ? 0 : 8;
  const float scale = 0.0625f;  // 256^-0.5

  v8f acc[16] = {};
  float m_run = -1e30f, l_run = 0.0f;

  for (int kv0 = c0 * 32; kv0 < c1 * 32; kv0 += 32) {
    if (kv0 + 32 < c1 * 32) {  // prefetch next K / V^T chunks (global_prefetch_b8)
      __builtin_prefetch((const void*)(k + (size_t)(kv0 + 32 + (lane & 15)) * DM), 0, 1);
      __builtin_prefetch((const void*)(vt + (size_t)(lane & 15) * vts + kv0 + 32), 0, 1);
    }

    // logits^T for two 16-key sub-tiles; each Q^T fragment feeds both
    v8f s0 = {}, s1 = {};
#pragma unroll
    for (int dc = 0; dc < 8; ++dc) {
      v16h bq = load_b_frag(q, DM, qcol, dc * 32, lane);
      v16h a0 = load_a_frag(k, DM, kv0 + (lane & 15), dc * 32, lane);
      v16h a1 = load_a_frag(k, DM, kv0 + 16 + (lane & 15), dc * 32, lane);
      s0 = WMMA_F16(a0, bq, s0);
      s1 = WMMA_F16(a1, bq, s1);
    }

    float sv0[8], sv1[8];
#pragma unroll
    for (int j = 0; j < 8; ++j) {
      int k0 = kv0 + krow + j;
      int k1 = kv0 + 16 + krow + j;
      sv0[j] = (k0 < skv) ? s0[j] * scale : -1e30f;
      sv1[j] = (k1 < skv) ? s1[j] * scale : -1e30f;
    }
    float mx = -1e30f;
#pragma unroll
    for (int j = 0; j < 8; ++j) mx = fmaxf(mx, fmaxf(sv0[j], sv1[j]));
    mx = fmaxf(mx, __shfl_xor(mx, 16, 32));
    float m_new = fmaxf(m_run, mx);
    float alpha = __expf(m_run - m_new);

    float p0[8], p1[8], ls = 0.0f;
#pragma unroll
    for (int j = 0; j < 8; ++j) {
      p0[j] = __expf(sv0[j] - m_new);
      p1[j] = __expf(sv1[j] - m_new);
      ls += p0[j] + p1[j];
    }
    ls += __shfl_xor(ls, 16, 32);
    l_run = l_run * alpha + ls;
    m_run = m_new;

#pragma unroll
    for (int dc = 0; dc < 16; ++dc)
#pragma unroll
      for (int j = 0; j < 8; ++j) acc[dc][j] *= alpha;

    // Build P^T B-fragment: exchange the opposite key-half with lane^16
    union PU { unsigned u[8]; _Float16 h[16]; } pu, qu;
#pragma unroll
    for (int j = 0; j < 8; ++j) {
      pu.h[j]     = (_Float16)p0[j];
      pu.h[8 + j] = (_Float16)p1[j];
    }
#pragma unroll
    for (int w = 0; w < 8; ++w)
      qu.u[w] = (unsigned)__shfl_xor((int)pu.u[w], 16, 32);
    v16h bp;
#pragma unroll
    for (int j = 0; j < 8; ++j) {
      bp[j]     = (lane < 16) ? pu.h[j] : qu.h[8 + j];
      bp[8 + j] = (lane < 16) ? qu.h[j] : pu.h[8 + j];
    }

    // O^T += V^T @ P^T over all 16 head-dim chunks
#pragma unroll
    for (int dc = 0; dc < 16; ++dc) {
      v16h av = load_a_frag(vt, vts, dc * 16 + (lane & 15), kv0, lane);
      acc[dc] = WMMA_F16(av, bp, acc[dc]);
    }
  }

  // locally-normalized partial (keeps f16 range), plus split stats
  float inv = 1.0f / l_run;
  _Float16* ob = opart + ((size_t)split * SQ + qcol) * DM;
#pragma unroll
  for (int dc = 0; dc < 16; ++dc) {
    v8h st;
#pragma unroll
    for (int j = 0; j < 8; ++j) st[j] = (_Float16)(acc[dc][j] * inv);
    *(v8h*)(ob + dc * 16 + krow) = st;
  }
  if (lane < 16) {
    mbuf[split * SQ + qcol] = m_run;
    lbuf[split * SQ + qcol] = l_run;
  }
}

// Merge the KV splits: out = sum_s exp(m_s-M)*l_s*opart_s / sum_s exp(m_s-M)*l_s
__global__ void k_attn_combine(const _Float16* __restrict__ opart,
                               const float* __restrict__ mbuf,
                               const float* __restrict__ lbuf,
                               _Float16* __restrict__ out) {
  int idx = blockIdx.x * blockDim.x + threadIdx.x;
  if (idx >= SQ * DM) return;
  int qrow = idx >> 8;
  float M = -1e30f;
#pragma unroll
  for (int s = 0; s < SPLITS; ++s) M = fmaxf(M, mbuf[s * SQ + qrow]);
  float den = 0.0f, num = 0.0f;
#pragma unroll
  for (int s = 0; s < SPLITS; ++s) {
    float w = __expf(mbuf[s * SQ + qrow] - M) * lbuf[s * SQ + qrow];
    den += w;
    num += w * (float)opart[(size_t)s * SQ * DM + idx];
  }
  out[idx] = (_Float16)(num / den);
}

// ---------------------------------------------------------------------------
// Host orchestration
// ---------------------------------------------------------------------------
extern "C" void kernel_launch(void* const* d_in, const int* in_sizes, int n_in,
                              void* d_out, int out_size, void* d_ws, size_t ws_size,
                              hipStream_t stream) {
  (void)in_sizes; (void)n_in; (void)out_size; (void)ws_size;
  const float* queries = (const float*)d_in[0];
  const float* keys    = (const float*)d_in[1];
  const float* kpe     = (const float*)d_in[2];
  const float* cosT    = (const float*)d_in[3];
  const float* sinT    = (const float*)d_in[4];
  const float* sa_q_w = (const float*)d_in[5];  const float* sa_q_b = (const float*)d_in[6];
  const float* sa_k_w = (const float*)d_in[7];  const float* sa_k_b = (const float*)d_in[8];
  const float* sa_v_w = (const float*)d_in[9];  const float* sa_v_b = (const float*)d_in[10];
  const float* sa_o_w = (const float*)d_in[11]; const float* sa_o_b = (const float*)d_in[12];
  const float* ca_q_w = (const float*)d_in[13]; const float* ca_q_b = (const float*)d_in[14];
  const float* ca_k_w = (const float*)d_in[15]; const float* ca_k_b = (const float*)d_in[16];
  const float* ca_v_w = (const float*)d_in[17]; const float* ca_v_b = (const float*)d_in[18];
  const float* ca_o_w = (const float*)d_in[19]; const float* ca_o_b = (const float*)d_in[20];
  const float* mlp1_w = (const float*)d_in[21]; const float* mlp1_b = (const float*)d_in[22];
  const float* mlp2_w = (const float*)d_in[23]; const float* mlp2_b = (const float*)d_in[24];
  const float* ln1_w = (const float*)d_in[25];  const float* ln1_b = (const float*)d_in[26];
  const float* ln2_w = (const float*)d_in[27];  const float* ln2_b = (const float*)d_in[28];
  const float* ln3_w = (const float*)d_in[29];  const float* ln3_b = (const float*)d_in[30];
  float* resid = (float*)d_out;

  _Float16* W = (_Float16*)d_ws;  // offsets in halves
  _Float16* saq_wt  = W + 0;
  _Float16* sak_wt  = W + 65536;
  _Float16* sav_wt  = W + 131072;
  _Float16* sao_wt  = W + 196608;
  _Float16* caq_wt  = W + 262144;
  _Float16* cao_wt  = W + 327680;
  _Float16* cak_wt  = W + 393216;   // [256 x 64]
  _Float16* cav_wt  = W + 409600;   // [256 x 64]
  _Float16* mlp1_wt = W + 425984;   // [2048 x 256]
  _Float16* mlp2_wt = W + 950272;   // [256 x 2048]
  _Float16* h16    = W + 1474560;   // [4096 x 256]
  _Float16* bufQ   = W + 2523136;   // [4096 x 256]
  _Float16* bufO   = W + 3571712;   // [4096 x 256]
  _Float16* kin16  = W + 4620288;   // [28704 x 64]
  _Float16* keys16 = W + 6457344;   // [28704 x 64]
  _Float16* bufK   = W + 8294400;   // [28704 x 256]
  _Float16* bufVt  = W + 15642624;  // [256 x 28704]
  _Float16* opart  = W + 22990848;  // [8 x 4096 x 256] f16 partial O
  float*    mbuf   = (float*)(W + 31379456);           // [8 x 4096]
  float*    lbuf   = (float*)(W + 31379456) + SPLITS * SQ;
  _Float16* t16    = bufK;          // [4096 x 2048] reuses dead K/Vt region

  // ---- weight transposes (f32 -> f16 W^T) ----
  k_wt<<<(DM*DM+255)/256,256,0,stream>>>(sa_q_w, saq_wt, DM, DM);
  k_wt<<<(DM*DM+255)/256,256,0,stream>>>(sa_k_w, sak_wt, DM, DM);
  k_wt<<<(DM*DM+255)/256,256,0,stream>>>(sa_v_w, sav_wt, DM, DM);
  k_wt<<<(DM*DM+255)/256,256,0,stream>>>(sa_o_w, sao_wt, DM, DM);
  k_wt<<<(DM*DM+255)/256,256,0,stream>>>(ca_q_w, caq_wt, DM, DM);
  k_wt<<<(KVD*DM+255)/256,256,0,stream>>>(ca_k_w, cak_wt, KVD, DM);
  k_wt<<<(KVD*DM+255)/256,256,0,stream>>>(ca_v_w, cav_wt, KVD, DM);
  k_wt<<<(DM*DM+255)/256,256,0,stream>>>(ca_o_w, cao_wt, DM, DM);
  k_wt<<<(DM*FFD+255)/256,256,0,stream>>>(mlp1_w, mlp1_wt, DM, FFD);
  k_wt<<<(FFD*DM+255)/256,256,0,stream>>>(mlp2_w, mlp2_wt, FFD, DM);

  // ---- residual stream ----
  k_copy_f32<<<(SQ*DM+255)/256,256,0,stream>>>(queries, resid, SQ*DM);

  // ---- stage 1: self-attention ----
  k_ln<<<SQ,256,0,stream>>>(resid, ln1_w, ln1_b, h16);
  k_gemm<0,false><<<256,128,0,stream>>>(h16, saq_wt, sa_q_b, bufQ, nullptr, SQ, DM, DM, 0);
  k_gemm<0,false><<<256,128,0,stream>>>(h16, sak_wt, sa_k_b, bufK, nullptr, SQ, DM, DM, 0);
  k_gemm<1,false><<<256,128,0,stream>>>(h16, sav_wt, sa_v_b, bufVt, nullptr, SQ, DM, DM, SQ);
  k_rope<<<SQ,128,0,stream>>>(bufQ, cosT, sinT, 0);
  k_rope<<<SQ,128,0,stream>>>(bufK, cosT, sinT, 0);
  k_attn<<<dim3(64,SPLITS),128,0,stream>>>(bufQ, bufK, bufVt, opart, mbuf, lbuf,
                                           SQ, SQ, /*nchunks=*/128, /*cpw=*/16);
  k_attn_combine<<<(SQ*DM+255)/256,256,0,stream>>>(opart, mbuf, lbuf, bufO);
  k_gemm<2,false><<<256,128,0,stream>>>(bufO, sao_wt, sa_o_b, nullptr, resid, SQ, DM, DM, 0);

  // ---- stage 2: cross-attention ----
  k_ln<<<SQ,256,0,stream>>>(resid, ln2_w, ln2_b, h16);
  k_addcvt<<<(SKVP*KVD+255)/256,256,0,stream>>>(keys, kpe, kin16, keys16);
  k_gemm<0,false><<<256,128,0,stream>>>(h16, caq_wt, ca_q_b, bufQ, nullptr, SQ, DM, DM, 0);
  k_gemm<0,false><<<1794,128,0,stream>>>(kin16, cak_wt, ca_k_b, bufK, nullptr, SKVP, DM, KVD, 0);
  k_gemm<1,false><<<1794,128,0,stream>>>(keys16, cav_wt, ca_v_b, bufVt, nullptr, SKVP, DM, KVD, SKVP);
  k_rope<<<SQ,128,0,stream>>>(bufQ, cosT, sinT, 0);
  k_rope<<<SKVP,128,0,stream>>>(bufK, cosT, sinT, 1);
  k_attn<<<dim3(64,SPLITS),128,0,stream>>>(bufQ, bufK, bufVt, opart, mbuf, lbuf,
                                           SKV, SKVP, /*nchunks=*/897, /*cpw=*/113);
  k_attn_combine<<<(SQ*DM+255)/256,256,0,stream>>>(opart, mbuf, lbuf, bufO);
  k_gemm<2,false><<<256,128,0,stream>>>(bufO, cao_wt, ca_o_b, nullptr, resid, SQ, DM, DM, 0);

  // ---- stage 3: MLP ----
  k_ln<<<SQ,256,0,stream>>>(resid, ln3_w, ln3_b, h16);
  k_gemm<0,true><<<2048,128,0,stream>>>(h16, mlp1_wt, mlp1_b, t16, nullptr, SQ, FFD, DM, 0);
  k_gemm<2,false><<<256,128,0,stream>>>(t16, mlp2_wt, mlp2_b, nullptr, resid, SQ, DM, FFD, 0);
}
